// IntegralOperator_91044716740913
// MI455X (gfx1250) — matrix operand order repaired and checked by reading, not compiled, not run on hardware
//
#include <hip/hip_runtime.h>

#define NN 1024
#define HH 64
#define DYC (1.0f / 1024.0f)

typedef __attribute__((ext_vector_type(16))) _Float16 v16h;
typedef __attribute__((ext_vector_type(8)))  float    v8f;

__device__ __forceinline__ float fast_tanh(float x) {
#if __has_builtin(__builtin_amdgcn_tanhf)
    return __builtin_amdgcn_tanhf(x);
#else
    // tanh(x) = 1 - 2/(exp(2x)+1), exp via native exp2
    float e = __builtin_amdgcn_exp2f(x * 2.8853900817779268f); // 2*log2(e)
    return 1.0f - 2.0f * __builtin_amdgcn_rcpf(e + 1.0f);
#endif
}

// Kernel 1: out[i] = hom[i] + DY * b3 * sum(f)   (pure store -> replay-safe)
__global__ __launch_bounds__(256)
void pino_hom_init(const float* __restrict__ f,  const float* __restrict__ ys,
                   const float* __restrict__ b3,
                   const float* __restrict__ Hw1, const float* __restrict__ Hb1,
                   const float* __restrict__ Hw2, const float* __restrict__ Hb2,
                   float* __restrict__ out) {
    int i = blockIdx.x * blockDim.x + threadIdx.x;
    if (i >= NN) return;
    float sumf = 0.f;
    for (int j = 0; j < NN; ++j) sumf += f[j];          // tiny, L2-resident
    float y = ys[i];
    float hom = Hb2[0];
    for (int n = 0; n < HH; ++n)
        hom = fmaf(fast_tanh(fmaf(y, Hw1[n], Hb1[n])), Hw2[n], hom);
    out[i] = hom + DYC * b3[0] * sumf;
}

// Kernel 2: each wave owns one i and a 256-wide j-chunk; 16 batches of 16 pairs.
// Per batch: layer1 in VALU (lane computes exactly its A-fragment elements),
// layer2 = 8x v_wmma_f32_16x16x32_f16, epilogue tanh + W3 dot + shuffle-reduce.
__global__ __launch_bounds__(256)
void pino_main(const float* __restrict__ f,  const float* __restrict__ ys,
               const float* __restrict__ W1, const float* __restrict__ b1,
               const float* __restrict__ W2, const float* __restrict__ b2,
               const float* __restrict__ W3, float* __restrict__ out) {
    const int lane  = threadIdx.x & 31;
    const int wave  = threadIdx.x >> 5;
    const int gw    = blockIdx.x * 8 + wave;   // 0..4095
    const int i     = gw >> 2;                 // 0..1023
    const int jchnk = gw & 3;
    const int jbase = jchnk * 256;

    const int lhalf = lane & 15;
    const int hi    = (lane >> 4) & 1;

    // ---- W2 B-fragments (constant): B[32x16 f16], lane holds N=lhalf,
    //      K_local = h + (h>=8?8:0) + 8*hi. Fragment (kt,nt) covers K=kt*32.., N=nt*16..
    v16h Bfrag[2][4];
#pragma unroll
    for (int kt = 0; kt < 2; ++kt)
#pragma unroll
        for (int nt = 0; nt < 4; ++nt)
#pragma unroll
            for (int h = 0; h < 16; ++h) {
                int k = kt * 32 + h + (h >= 8 ? 8 : 0) + hi * 8;
                int n = nt * 16 + lhalf;
                Bfrag[kt][nt][h] = (_Float16)W2[k * HH + n];
            }

    // ---- layer-1 per-lane constants: A-fragment half h maps to feature
    //      K = h + (h>=8?8:0) + 8*hi (+32*kt)
    const float ysi = ys[i];
    float pre[32], w1y[32];
#pragma unroll
    for (int kt = 0; kt < 2; ++kt)
#pragma unroll
        for (int h = 0; h < 16; ++h) {
            int k = kt * 32 + h + (h >= 8 ? 8 : 0) + hi * 8;
            pre[kt * 16 + h] = fmaf(ysi, W1[0 * HH + k], b1[k]);
            w1y[kt * 16 + h] = W1[1 * HH + k];
        }

    // ---- epilogue constants for this lane's N columns
    float b2v[4], w3v[4];
#pragma unroll
    for (int nt = 0; nt < 4; ++nt) {
        b2v[nt] = b2[nt * 16 + lhalf];
        w3v[nt] = W3[nt * 16 + lhalf];
    }

    const bool redlane = (lhalf < 8);
    const int  midx    = lhalf + hi * 8;   // row M this lane accumulates (if redlane)

    float acc = 0.f;

#pragma unroll 1
    for (int b = 0; b < 16; ++b) {
        const int j0  = jbase + b * 16;
        const float ysj = ys[j0 + lhalf];          // this lane's A rows have M = lhalf
        const float fsv = f[j0 + (redlane ? midx : 0)];

        // layer 1 -> A fragments (f16)
        v16h A[2];
#pragma unroll
        for (int kt = 0; kt < 2; ++kt)
#pragma unroll
            for (int h = 0; h < 16; ++h)
                A[kt][h] = (_Float16)fast_tanh(fmaf(ysj, w1y[kt * 16 + h], pre[kt * 16 + h]));

        // layer 2: (16x64) @ (64x64) via 8 WMMAs
        v8f C[4];
#pragma unroll
        for (int nt = 0; nt < 4; ++nt) {
            v8f c = {0.f, 0.f, 0.f, 0.f, 0.f, 0.f, 0.f, 0.f};
            c = __builtin_amdgcn_wmma_f32_16x16x32_f16(false, A[0], false, Bfrag[0][nt],
                                                       (short)0, c, false, false);
            c = __builtin_amdgcn_wmma_f32_16x16x32_f16(false, A[1], false, Bfrag[1][nt],
                                                       (short)0, c, false, false);
            C[nt] = c;
        }

        // epilogue: tanh(+b2), partial dot with W3 (lane covers 4 of 64 columns)
        float p[8];
#pragma unroll
        for (int r = 0; r < 8; ++r) p[r] = 0.f;
#pragma unroll
        for (int nt = 0; nt < 4; ++nt)
#pragma unroll
            for (int r = 0; r < 8; ++r)
                p[r] = fmaf(fast_tanh(C[nt][r] + b2v[nt]), w3v[nt], p[r]);

        // reduce each row's dot over its 16-lane group; weight by f[j], accumulate
#pragma unroll
        for (int r = 0; r < 8; ++r) {
            float d = p[r];
            d += __shfl_xor(d, 1);
            d += __shfl_xor(d, 2);
            d += __shfl_xor(d, 4);
            d += __shfl_xor(d, 8);
            if (redlane && (lhalf == r)) acc = fmaf(d, fsv, acc);
        }
    }

    // wave-wide reduction, one atomic per wave
    acc += __shfl_xor(acc, 1);
    acc += __shfl_xor(acc, 2);
    acc += __shfl_xor(acc, 4);
    acc += __shfl_xor(acc, 8);
    acc += __shfl_xor(acc, 16);
    if (lane == 0) atomicAdd(&out[i], acc * DYC);
}

extern "C" void kernel_launch(void* const* d_in, const int* in_sizes, int n_in,
                              void* d_out, int out_size, void* d_ws, size_t ws_size,
                              hipStream_t stream) {
    const float* f   = (const float*)d_in[0];
    const float* ys  = (const float*)d_in[1];
    const float* W1  = (const float*)d_in[2];
    const float* b1  = (const float*)d_in[3];
    const float* W2  = (const float*)d_in[4];
    const float* b2  = (const float*)d_in[5];
    const float* W3  = (const float*)d_in[6];
    const float* b3  = (const float*)d_in[7];
    const float* Hw1 = (const float*)d_in[8];
    const float* Hb1 = (const float*)d_in[9];
    const float* Hw2 = (const float*)d_in[10];
    const float* Hb2 = (const float*)d_in[11];
    float* out = (float*)d_out;

    // Pass 1: out[i] = hom[i] + DY*b3*sum(f)   (overwrite -> deterministic replays)
    pino_hom_init<<<NN / 256, 256, 0, stream>>>(f, ys, b3, Hw1, Hb1, Hw2, Hb2, out);

    // Pass 2: 4096 waves (1024 i x 4 j-chunks), atomic accumulate integral term
    pino_main<<<512, 256, 0, stream>>>(f, ys, W1, b1, W2, b2, W3, out);
}
